// DnTLayer_17995912970777
// MI455X (gfx1250) — matrix operand order repaired and checked by reading, not compiled, not run on hardware
//
#include <hip/hip_runtime.h>
#include <hip/hip_bf16.h>

typedef __attribute__((ext_vector_type(16))) __bf16 v16bf_t;
typedef __attribute__((ext_vector_type(8)))  __bf16 v8bf_t;
typedef __attribute__((ext_vector_type(8)))  float  v8f_t;
typedef __attribute__((ext_vector_type(4)))  unsigned int v4u_t;
typedef __attribute__((ext_vector_type(8)))  int v8i_t;
typedef __attribute__((ext_vector_type(4)))  int v4i_t;

#define B_  128
#define S_  19
#define HW_ (S_*S_)

// ---------------- zero the padded bf16 activation buffer ----------------
__global__ void k_zero16(uint4* __restrict__ p, long n) {
  long i = (long)blockIdx.x * blockDim.x + threadIdx.x;
  if (i < n) p[i] = make_uint4(0u, 0u, 0u, 0u);
}

// ------- fused nearest-resize + 9x9 point correlation (one dy per blockIdx.y) -------
template<int C, int HS>
__global__ void k_corr(const float* __restrict__ t0, const float* __restrict__ t1,
                       __bf16* __restrict__ act, int ch_off) {
  int idx = blockIdx.x * blockDim.x + threadIdx.x;
  if (idx >= B_ * HW_) return;
  int dy = blockIdx.y;                     // 0..8
  int b = idx / HW_, p = idx % HW_;
  int y = p / S_, x = p % S_;
  int sy0 = (y * HS) / S_, sx0 = (x * HS) / S_;
  int yy = y + dy - 4;
  bool rowin = (yy >= 0) && (yy < S_);
  float acc[9] = {0.f,0.f,0.f,0.f,0.f,0.f,0.f,0.f,0.f};
  if (rowin) {
    int sy1 = (yy * HS) / S_;
    int sx1[9]; bool vx[9];
#pragma unroll
    for (int j = 0; j < 9; ++j) {
      int xx = x + j - 4;
      vx[j] = (xx >= 0) && (xx < S_);
      sx1[j] = vx[j] ? (xx * HS) / S_ : 0;
    }
    constexpr long chst = (long)HS * HS;
    const float* p0 = t0 + ((size_t)b * C * HS + sy0) * HS + sx0;
    const float* p1 = t1 + ((size_t)b * C * HS + sy1) * HS;
#pragma unroll 4
    for (int c = 0; c < C; ++c) {
      float a = p0[c * chst];
      const float* r1 = p1 + c * chst;
#pragma unroll
      for (int j = 0; j < 9; ++j)
        if (vx[j]) acc[j] = fmaf(a, r1[sx1[j]], acc[j]);
    }
  }
  __bf16* o = act + (size_t)idx * 256 + ch_off + dy * 9;
#pragma unroll
  for (int j = 0; j < 9; ++j) o[j] = (__bf16)acc[j];
}

// ------- repack f32 OIHW weights -> bf16 [oc][tap][ICpad] -------
__global__ void k_repack(const float* __restrict__ w, __bf16* __restrict__ wr,
                         int OC, int IC, int ICp, int T) {
  int i = blockIdx.x * blockDim.x + threadIdx.x;
  if (i >= OC * T * ICp) return;
  int ic = i % ICp;
  int t  = (i / ICp) % T;
  int oc = i / (ICp * T);
  float v = (ic < IC) ? w[((size_t)oc * IC + ic) * T + t] : 0.f;
  wr[i] = (__bf16)v;
}

// A-fragment loader: 16-bit A-matrix 16x32 layout (ISA 7.12.2).
__device__ __forceinline__ v16bf_t ldA(const __bf16* p, int hl) {
  v8bf_t lo = *(const v8bf_t*)(p + hl * 8);
  v8bf_t hi = *(const v8bf_t*)(p + hl * 8 + 16);
  return __builtin_shufflevector(lo, hi, 0,1,2,3,4,5,6,7,8,9,10,11,12,13,14,15);
}

__device__ __forceinline__ v8f_t wmma_bf16(v16bf_t a, v16bf_t b, v8f_t c) {
  return __builtin_amdgcn_wmma_f32_16x16x32_bf16(false, a, false, b, (short)0, c,
                                                 false, false);
}

// ---- TDM: DMA one weight-tap slab (32 rows x ICp bf16, row stride T*ICp) into
// LDS with hardware row padding (+16B) for bank rotation.  D# per ISA 8.3/8.4.
template<int ICp, int T>
__device__ __forceinline__ void tdm_stage_weights(const __bf16* gsrc, unsigned lds_off) {
  constexpr unsigned PI = (ICp == 128) ? 5u : 6u;  // pad interval: 8B*2^PI == row bytes
  unsigned long long ga = (unsigned long long)(size_t)gsrc;
  v4u_t g0;
  g0[0] = 1u;                                            // count=1 (valid D#)
  g0[1] = lds_off;                                       // lds_addr (bytes)
  g0[2] = (unsigned)(ga & 0xFFFFFFFFull);                // global_addr[31:0]
  g0[3] = (unsigned)((ga >> 32) & 0x1FFFFFFull) | (2u << 30);  // ga[56:32] | type=2
  v8i_t g1;
  g1[0] = (int)((1u << 16)            // data_size = 2 bytes
              | (1u << 20)            // pad_enable
              | (PI << 22)            // pad_interval
              | (3u << 25));          // pad_amount = 4 DWORDs (16 B)
  g1[1] = (int)(((unsigned)ICp & 0xFFFFu) << 16);        // tensor_dim0[15:0] @ bits63:48
  g1[2] = (int)(32u << 16);                              // tensor_dim1=32 @ bits95:80 (dim0 hi = 0)
  g1[3] = (int)((unsigned)ICp << 16);                    // tile_dim0 @ bits127:112
  g1[4] = 32;                                            // tile_dim1=32; tile_dim2=0
  g1[5] = T * ICp;                                       // tensor_dim0_stride[31:0]
  g1[6] = 0;                                             // stride hi / dim1_stride lo
  g1[7] = 0;
  v4i_t gz = {};
#if defined(__clang_major__) && (__clang_major__ >= 23)
  v8i_t gz8 = {};
  __builtin_amdgcn_tensor_load_to_lds(g0, g1, gz, gz, gz8, 0);
#else
  __builtin_amdgcn_tensor_load_to_lds(g0, g1, gz, gz, 0);
#endif
}

// ------- implicit-GEMM conv via V_WMMA_F32_16X16X32_BF16 -------
// One wave computes a 32x32 tile; 4 waves/block share 32 weight rows that the
// TDM streams tap-by-tap into double-buffered, row-padded LDS while WMMAs run.
template<int ICp, int K3, int STRIDE, int HIN, int HOUT, int OC>
__global__ void __launch_bounds__(128)
k_conv_wmma(const __bf16* __restrict__ act_in, const __bf16* __restrict__ wrep,
            const float* __restrict__ bias, float* __restrict__ out_f32,
            __bf16* __restrict__ act_out)
{
  constexpr int T    = K3 * K3;
  constexpr int PAD  = (K3 == 3) ? 1 : 0;
  constexpr int HWo  = HOUT * HOUT;
  constexpr int NB   = (T > 1) ? 2 : 1;     // LDS tap buffers
  constexpr int LROW = ICp + 8;             // TDM-padded LDS row (bank rotation)
  constexpr int SLAB = 32 * LROW;           // elements per tap slab

  __shared__ __bf16 ldsw[NB * SLAB];

  const int lane = threadIdx.x & 31;
  const int wv   = threadIdx.x >> 5;
  const int hl   = lane >> 4;               // K-half select (wave32 WMMA layout)
  const int l15  = lane & 15;
  const int n0   = (blockIdx.x * 4 + wv) * 32;
  const int m0   = blockIdx.y * 32;

  auto stage = [&](int tap, int buf) {      // wave 0 only; TENSORcnt-tracked DMA
    const __bf16* src = wrep + (size_t)m0 * T * ICp + (size_t)tap * ICp;
    tdm_stage_weights<ICp, T>(src, (unsigned)(size_t)&ldsw[buf * SLAB]);
  };

  int col[2], bb[2], pp[2], oy[2], ox[2];
#pragma unroll
  for (int t = 0; t < 2; ++t) {
    col[t] = n0 + 16 * t + l15;
    bb[t] = col[t] / HWo;  pp[t] = col[t] % HWo;   // constant divisor
    oy[t] = pp[t] / HOUT;  ox[t] = pp[t] % HOUT;
  }

  if (wv == 0) {
    stage(0, 0);
    __builtin_amdgcn_s_wait_tensorcnt(0);
  }
  __syncthreads();

  v8f_t c00 = {}, c01 = {}, c10 = {}, c11 = {};   // c[mh][nh]
  for (int tap = 0; tap < T; ++tap) {
    const int cb = tap % NB;
    const bool pend = (T > 1) && (tap + 1 < T);
    if (pend && wv == 0) stage(tap + 1, (tap + 1) % NB);

    const int ky = tap / K3, kx = tap % K3;
    bool inb[2]; size_t boff[2];
#pragma unroll
    for (int t = 0; t < 2; ++t) {
      int iy = oy[t] * STRIDE + ky - PAD;
      int ix = ox[t] * STRIDE + kx - PAD;
      inb[t]  = (iy >= 0) && (iy < HIN) && (ix >= 0) && (ix < HIN);
      boff[t] = inb[t] ? ((size_t)(bb[t] * HIN * HIN + iy * HIN + ix)) * ICp : 0;
      if (inb[t]) __builtin_prefetch(act_in + boff[t], 0, 0);  // global_prefetch_b8
    }
    const __bf16* a0 = &ldsw[cb * SLAB + l15 * LROW];          // A rows via LDS
    const __bf16* a1 = a0 + 16 * LROW;
#pragma unroll
    for (int k = 0; k < ICp; k += 32) {
      v16bf_t A0 = ldA(a0 + k, hl);
      v16bf_t A1 = ldA(a1 + k, hl);
      v16bf_t B0 = {}, B1 = {};
      if (inb[0]) B0 = ldA(act_in + boff[0] + k, hl);
      if (inb[1]) B1 = ldA(act_in + boff[1] + k, hl);
      c00 = wmma_bf16(A0, B0, c00);
      c01 = wmma_bf16(A0, B1, c01);
      c10 = wmma_bf16(A1, B0, c10);
      c11 = wmma_bf16(A1, B1, c11);
    }
    if (pend && wv == 0) __builtin_amdgcn_s_wait_tensorcnt(0);  // DMA done...
    __syncthreads();                                            // ...publish buffer
  }

  // D layout: lane-half hl holds rows M = base + 8*hl + v, column = lane&15.
  v8f_t* cc[2][2] = {{&c00, &c01}, {&c10, &c11}};
#pragma unroll
  for (int mh = 0; mh < 2; ++mh) {
    const int ocb = m0 + 16 * mh + hl * 8;
    float bv[8];
#pragma unroll
    for (int v = 0; v < 8; ++v) bv[v] = bias[ocb + v];
#pragma unroll
    for (int nh = 0; nh < 2; ++nh) {
      v8f_t cv = *cc[mh][nh];
      float vals[8];
#pragma unroll
      for (int v = 0; v < 8; ++v) {
        float f = cv[v] + bv[v];
        vals[v] = f > 0.f ? f : 0.f;            // ReLU
      }
      if (out_f32) {
#pragma unroll
        for (int v = 0; v < 8; ++v)
          out_f32[((size_t)bb[nh] * OC + ocb + v) * HWo + pp[nh]] = vals[v];
      }
      if (act_out) {
        v8bf_t ob;
#pragma unroll
        for (int v = 0; v < 8; ++v) ob[v] = (__bf16)vals[v];
        *(v8bf_t*)(act_out + (size_t)col[nh] * OC + ocb) = ob;
      }
    }
  }
}

// ---------------------------------------------------------------------------
extern "C" void kernel_launch(void* const* d_in, const int* in_sizes, int n_in,
                              void* d_out, int out_size, void* d_ws, size_t ws_size,
                              hipStream_t stream)
{
  (void)in_sizes; (void)n_in; (void)out_size; (void)ws_size;

  const float* c3_t0 = (const float*)d_in[0];
  const float* c3_t1 = (const float*)d_in[1];
  const float* c4_t0 = (const float*)d_in[2];
  const float* c4_t1 = (const float*)d_in[3];
  const float* c5_t0 = (const float*)d_in[4];
  const float* c5_t1 = (const float*)d_in[5];
  const float* base_w1 = (const float*)d_in[6];
  const float* base_b1 = (const float*)d_in[7];
  const float* base_w2 = (const float*)d_in[8];
  const float* base_b2 = (const float*)d_in[9];
  const float *ex_w1[5], *ex_b1[5], *ex_w2[5], *ex_b2[5];
  for (int i = 0; i < 5; ++i) {
    ex_w1[i] = (const float*)d_in[10 + 4*i];
    ex_b1[i] = (const float*)d_in[11 + 4*i];
    ex_w2[i] = (const float*)d_in[12 + 4*i];
    ex_b2[i] = (const float*)d_in[13 + 4*i];
  }
  float* dout = (float*)d_out;
  char*  ws   = (char*)d_ws;

  const size_t N0  = (size_t)B_ * HW_;       // 46208 pixels at 19x19
  const size_t szA = N0 * 256 * 2;
  const size_t szB = N0 * 128 * 2;
  __bf16* actA = (__bf16*)(ws);
  __bf16* actB = (__bf16*)(ws + szA);
  size_t cur = szA + szB;
  auto wsalloc = [&](size_t elems) -> __bf16* {
    __bf16* r = (__bf16*)(ws + cur); cur += elems * 2; return r;
  };
  __bf16* wrB1 = wsalloc((size_t)128 * 256);        // 243 -> 256 padded
  __bf16* wrB2 = wsalloc((size_t)256 * 9 * 128);
  __bf16 *wrE1[5], *wrE2[5];
  for (int i = 0; i < 5; ++i) {
    wrE1[i] = wsalloc((size_t)128 * 256);
    wrE2[i] = wsalloc((size_t)256 * 9 * 128);
  }

  // 1) zero ACT0 (covers the 13 zero-pad channels 243..255)
  long n16 = (long)(szA / 16);
  k_zero16<<<dim3((unsigned)((n16 + 255) / 256)), dim3(256), 0, stream>>>((uint4*)actA, n16);

  // 2) resize + correlation -> ACT0 bf16 [n][256]
  dim3 cg((unsigned)((B_ * HW_ + 255) / 256), 9);
  k_corr<  32, 38><<<cg, 256, 0, stream>>>(c3_t0, c3_t1, actA,   0);
  k_corr<  96, 19><<<cg, 256, 0, stream>>>(c4_t0, c4_t1, actA,  81);
  k_corr<1280, 10><<<cg, 256, 0, stream>>>(c5_t0, c5_t1, actA, 162);

  // 3) repack weights to bf16 WMMA-friendly layout
  auto rp = [&](const float* w, __bf16* wr, int OC, int IC, int ICp, int T) {
    int tot = OC * T * ICp;
    k_repack<<<dim3((unsigned)((tot + 255) / 256)), dim3(256), 0, stream>>>(w, wr, OC, IC, ICp, T);
  };
  rp(base_w1, wrB1, 128, 243, 256, 1);
  rp(base_w2, wrB2, 256, 128, 128, 9);
  for (int i = 0; i < 5; ++i) {
    rp(ex_w1[i], wrE1[i], 128, 256, 256, 1);
    rp(ex_w2[i], wrE2[i], 256, 128, 128, 9);
  }

  // 4) conv stack. Block = 4 waves x (32x32 tile) = 128 columns.
  size_t o1 = (size_t)B_*256*361, o2 = o1 + (size_t)B_*256*100,
         o3 = o2 + (size_t)B_*256*25, o4 = o3 + (size_t)B_*256*9,
         o5 = o4 + (size_t)B_*256*4;

  k_conv_wmma<256,1,1,19,19,128><<<dim3(361,4), 128, 0, stream>>>(actA, wrB1, base_b1, nullptr, actB);
  k_conv_wmma<128,3,1,19,19,256><<<dim3(361,8), 128, 0, stream>>>(actB, wrB2, base_b2, dout,    actA);

  k_conv_wmma<256,1,1,19,19,128><<<dim3(361,4), 128, 0, stream>>>(actA, wrE1[0], ex_b1[0], nullptr,   actB);
  k_conv_wmma<128,3,2,19,10,256><<<dim3(100,8), 128, 0, stream>>>(actB, wrE2[0], ex_b2[0], dout + o1, actA);

  k_conv_wmma<256,1,1,10,10,128><<<dim3(100,4), 128, 0, stream>>>(actA, wrE1[1], ex_b1[1], nullptr,   actB);
  k_conv_wmma<128,3,2,10, 5,256><<<dim3( 25,8), 128, 0, stream>>>(actB, wrE2[1], ex_b2[1], dout + o2, actA);

  k_conv_wmma<256,1,1, 5, 5,128><<<dim3( 25,4), 128, 0, stream>>>(actA, wrE1[2], ex_b1[2], nullptr,   actB);
  k_conv_wmma<128,3,2, 5, 3,256><<<dim3(  9,8), 128, 0, stream>>>(actB, wrE2[2], ex_b2[2], dout + o3, actA);

  k_conv_wmma<256,1,1, 3, 3,128><<<dim3(  9,4), 128, 0, stream>>>(actA, wrE1[3], ex_b1[3], nullptr,   actB);
  k_conv_wmma<128,3,2, 3, 2,256><<<dim3(  4,8), 128, 0, stream>>>(actB, wrE2[3], ex_b2[3], dout + o4, actA);

  k_conv_wmma<256,1,1, 2, 2,128><<<dim3(  4,4), 128, 0, stream>>>(actA, wrE1[4], ex_b1[4], nullptr,   actB);
  k_conv_wmma<128,3,2, 2, 1,256><<<dim3(  1,8), 128, 0, stream>>>(actB, wrE2[4], ex_b2[4], dout + o5, actA);
}
